// Encoder_20675972563499
// MI455X (gfx1250) — matrix operand order repaired and checked by reading, not compiled, not run on hardware
//
#include <hip/hip_runtime.h>
#include <hip/hip_bf16.h>

typedef __attribute__((ext_vector_type(16))) __bf16 v16bf;
typedef __attribute__((ext_vector_type(8)))  __bf16 v8bf;
typedef __attribute__((ext_vector_type(8)))  float  v8f;

#define TSTEPS 64
#define NDIM   128
#define HDIM   128
#define GDIM   512
#define BT     16
#define OUT1OFF (1024ull*64ull*128ull)

// ---------------------------------------------------------------------------
// Prep kernel: pack [W_ih;W_hh] -> bf16 in WMMA B-matrix lane layout,
// precompute xw1/xw2 = X^T @ w_x, yterm = y_prev @ w2_y, summed biases.
// ---------------------------------------------------------------------------
__global__ __launch_bounds__(256) void prep_kernel(
    const float* __restrict__ X, const float* __restrict__ y_prev,
    const float* __restrict__ W_ih1, const float* __restrict__ W_hh1,
    const float* __restrict__ b_ih1, const float* __restrict__ b_hh1,
    const float* __restrict__ W_ih2, const float* __restrict__ W_hh2,
    const float* __restrict__ b_ih2, const float* __restrict__ b_hh2,
    const float* __restrict__ attn1_W, const float* __restrict__ attn2_W,
    __bf16* __restrict__ pW1, __bf16* __restrict__ pW2,
    float* __restrict__ xw1, float* __restrict__ xw2,
    float* __restrict__ yterm, float* __restrict__ bsum1, float* __restrict__ bsum2)
{
  int idx = blockIdx.x * 256 + threadIdx.x;
  if (idx < 262144) {
    // Weight packing. Element (which, tile=(nt*8+kt), lane L, i) holds
    // Wc[kt*32 + (L/16)*16 + i, nt*16 + L%16]  (B-matrix 32x16 bf16, contiguous k order matching the A loads).
    int which = idx >> 17;
    int rem   = idx & 131071;
    int tile  = rem >> 9;
    int w     = rem & 511;
    int L = w >> 4, i = w & 15;
    int nt = tile >> 3, kt = tile & 7;
    int col = nt * 16 + (L & 15);
    int kk  = kt * 32 + (L >> 4) * 16 + i;
    const float* Wi = which ? W_ih2 : W_ih1;
    const float* Wh = which ? W_hh2 : W_hh1;
    float v = (kk < 128) ? Wi[col * 128 + kk] : Wh[col * 128 + (kk - 128)];
    (which ? pW2 : pW1)[rem] = (__bf16)v;
  } else if (idx < 524288) {
    int r = idx - 262144;
    int which = r >> 17;
    int q = r & 131071;
    int b = q >> 7, n = q & 127;
    const float* aw = which ? attn2_W : attn1_W;
    float s = 0.f;
    for (int t = 0; t < TSTEPS; ++t)
      s += X[((size_t)b * TSTEPS + t) * NDIM + n] * aw[256 + t];
    (which ? xw2 : xw1)[q] = s;
  } else if (idx < 525312) {
    int b = idx - 524288;
    float s = 0.f;
    for (int t = 0; t < TSTEPS; ++t)
      s += y_prev[b * TSTEPS + t] * attn2_W[320 + t];
    yterm[b] = s;
  } else if (idx < 526336) {
    int j = idx - 525312;
    if (j < 512) bsum1[j] = b_ih1[j] + b_hh1[j];
    else         bsum2[j - 512] = b_ih2[j - 512] + b_hh2[j - 512];
  }
}

// ---------------------------------------------------------------------------
// Wave32 helpers
// ---------------------------------------------------------------------------
__device__ __forceinline__ float wred_sum(float v) {
#pragma unroll
  for (int m = 16; m >= 1; m >>= 1) v += __shfl_xor(v, m, 32);
  return v;
}
__device__ __forceinline__ float wred_max(float v) {
#pragma unroll
  for (int m = 16; m >= 1; m >>= 1) v = fmaxf(v, __shfl_xor(v, m, 32));
  return v;
}
__device__ __forceinline__ float sigf(float x) { return 1.f / (1.f + __expf(-x)); }

// ---------------------------------------------------------------------------
// Fused GEMM + LSTM cell. Wave w owns hidden columns [16w, 16w+16) of ALL
// FOUR gates (N-tiles {w, w+8, w+16, w+24}), so i/f/g/o for one (row,col)
// land in the same lane & accumulator index -> cell update straight out of
// the WMMA accumulators, no LDS gate staging.
// Loop order: kt outer / gate inner => one live A-frag, adjacent WMMAs hit
// different accumulators (no back-to-back same-D dependency), ~100 VGPRs.
// ---------------------------------------------------------------------------
__device__ __forceinline__ void gemm_cell(
    const __bf16* __restrict__ pW, const __bf16* actS,
    const float* bias4,            // preloaded per-lane biases for i,f,g,o
    float* hS, float* cS, __bf16* actHdst, int wave, int lane)
{
  const int half = lane >> 4;
  const int r    = lane & 15;
  v8f acc[4];
#pragma unroll
  for (int g = 0; g < 4; ++g) acc[g] = (v8f){};

  __builtin_prefetch((const char*)pW + wave * (8 * 512 * 2) + lane * 32, 0, 1);

#pragma unroll
  for (int kt = 0; kt < 8; ++kt) {
    // A-matrix 16x32 bf16, contiguous k order matching packed B: half selects K 0-15 vs 16-31
    const __bf16* p = actS + r * 256 + kt * 32 + half * 16;
    union { v16bf v; v8bf h[2]; } u;
    u.h[0] = *(const v8bf*)(p);
    u.h[1] = *(const v8bf*)(p + 8);
    const v16bf afrag = u.v;
#pragma unroll
    for (int g = 0; g < 4; ++g) {
      // Byte offset of B-frag for (gate g tile, kt); lane-contiguous 32B.
      int off = (((wave + g * 8) * 8 + kt) * 512 + lane * 16) * 2;
      // Opaque offset: stops LICM hoisting the loop-invariant weight loads
      // out of the 64-step time loop (spills), while keeping the base
      // pointer's global address space (=> global_load_b128, not flat).
      asm volatile("" : "+v"(off));
      const v16bf bfrag = *(const v16bf*)((const char*)pW + off);
      acc[g] = __builtin_amdgcn_wmma_f32_16x16x32_bf16(
          false, afrag, false, bfrag, (short)0, acc[g], false, false);
    }
  }
  // C/D layout: VGPR j, lanes 0-15 -> M=j, lanes 16-31 -> M=j+8; N=lane%16
  const int hc = wave * 16 + (lane & 15);
  const int rb = half * 8;
#pragma unroll
  for (int j = 0; j < 8; ++j) {
    const int row = rb + j;
    const float ig = acc[0][j] + bias4[0];
    const float fg = acc[1][j] + bias4[1];
    const float gg = acc[2][j] + bias4[2];
    const float og = acc[3][j] + bias4[3];
    const float cc = sigf(fg) * cS[row * HDIM + hc] + sigf(ig) * tanhf(gg);
    const float hh = sigf(og) * tanhf(cc);
    cS[row * HDIM + hc] = cc;
    hS[row * HDIM + hc] = hh;
    actHdst[row * 256 + 128 + hc] = (__bf16)hh;  // bf16 h for next step's A
  }
}

// ---------------------------------------------------------------------------
// Attention + softmax over N=128 for 2 batch rows per wave.
// LAYER==1: x source = X[b, t, :];  LAYER==2: x source = x_tilde (LDS).
// ---------------------------------------------------------------------------
template <int LAYER>
__device__ __forceinline__ void attention(
    int wave, int lane, int t, int b0,
    const float* __restrict__ Xg,
    const float* hS, const float* cS, const float* xwS,
    const float* wv_h, const float* wv_s, float bias_scalar,
    const float* ytS, const float* xsrcS,
    float* xoutF, __bf16* actS)
{
#pragma unroll
  for (int rr = 0; rr < 2; ++rr) {
    const int b = wave * 2 + rr;
    float part = 0.f;
#pragma unroll
    for (int q = 0; q < 4; ++q) {
      int k = lane * 4 + q;
      part += hS[b * HDIM + k] * wv_h[k] + cS[b * HDIM + k] * wv_s[k];
    }
    float a = wred_sum(part) + bias_scalar;
    if (LAYER == 2) a += ytS[b];
    float e[4], m = -3.4e38f;
#pragma unroll
    for (int q = 0; q < 4; ++q) {
      e[q] = a + xwS[b * NDIM + lane * 4 + q];
      m = fmaxf(m, e[q]);
    }
    m = wred_max(m);
    float p[4], s = 0.f;
#pragma unroll
    for (int q = 0; q < 4; ++q) { p[q] = __expf(e[q] - m); s += p[q]; }
    s = wred_sum(s);
    const float inv = 1.f / s;
#pragma unroll
    for (int q = 0; q < 4; ++q) {
      const int n = lane * 4 + q;
      float xv;
      if (LAYER == 1)
        xv = Xg[((size_t)(b0 + b) * TSTEPS + t) * NDIM + n];
      else
        xv = xsrcS[b * NDIM + n];
      const float xt = p[q] * inv * xv;
      xoutF[b * NDIM + n] = xt;
      actS[b * 256 + n]   = (__bf16)xt;
    }
  }
}

// ---------------------------------------------------------------------------
// Persistent main kernel: one WG (8 wave32s) per 16 batch rows, loops all
// 64 timesteps internally (recurrence is independent per batch row).
// ---------------------------------------------------------------------------
__global__ __launch_bounds__(256) void encoder_main(
    const float* __restrict__ X,
    const __bf16* __restrict__ pW1, const __bf16* __restrict__ pW2,
    const float* __restrict__ xw1, const float* __restrict__ xw2,
    const float* __restrict__ yterm,
    const float* __restrict__ bsum1, const float* __restrict__ bsum2,
    const float* __restrict__ attn1_W, const float* __restrict__ attn1_b,
    const float* __restrict__ attn2_W, const float* __restrict__ attn2_b,
    float* __restrict__ out)
{
  __shared__ __bf16 actA1[BT * 256];      // [x_tilde1 | h1] bf16
  __shared__ __bf16 actA2[BT * 256];      // [x_tilde2 | h2] bf16
  __shared__ float  h1s[BT * HDIM], c1s[BT * HDIM];
  __shared__ float  h2s[BT * HDIM], c2s[BT * HDIM];
  __shared__ float  xtF[BT * NDIM];       // x_tilde1 f32
  __shared__ float  xt2F[BT * NDIM];      // x_tilde2 f32 (output 0)
  __shared__ float  xw1s[BT * NDIM], xw2s[BT * NDIM];
  __shared__ float  wv[512];              // w1_h w1_s w2_h w2_s
  __shared__ float  bsumS[2 * GDIM];
  __shared__ float  ytS[BT];
  __shared__ float  scal[2];

  const int tid  = threadIdx.x;
  const int wave = tid >> 5;
  const int lane = tid & 31;
  const int b0   = blockIdx.x * BT;

  // ---- init ----
  wv[tid]       = attn1_W[tid];           // 256 threads cover [0,256)
  wv[256 + tid] = attn2_W[tid];
  for (int i = tid; i < GDIM; i += 256) {
    bsumS[i]        = bsum1[i];
    bsumS[GDIM + i] = bsum2[i];
  }
  for (int i = tid; i < BT * NDIM; i += 256) {
    int b = i >> 7, n = i & 127;
    xw1s[i] = xw1[(size_t)(b0 + b) * NDIM + n];
    xw2s[i] = xw2[(size_t)(b0 + b) * NDIM + n];
    h1s[i] = 0.f; c1s[i] = 0.f; h2s[i] = 0.f; c2s[i] = 0.f;
    actA1[b * 256 + 128 + n] = (__bf16)0.f;
    actA2[b * 256 + 128 + n] = (__bf16)0.f;
  }
  if (tid < BT) ytS[tid] = yterm[b0 + tid];
  if (tid == 0) { scal[0] = attn1_b[0]; scal[1] = attn2_b[0]; }
  __syncthreads();

  // Per-lane gate biases (i,f,g,o at this lane's fixed hidden column).
  float bias1r[4], bias2r[4];
  {
    const int hc = wave * 16 + (lane & 15);
#pragma unroll
    for (int g = 0; g < 4; ++g) {
      bias1r[g] = bsumS[g * 128 + hc];
      bias2r[g] = bsumS[GDIM + g * 128 + hc];
    }
  }

  for (int t = 0; t < TSTEPS; ++t) {
    // layer 1 attention -> x_tilde1
    attention<1>(wave, lane, t, b0, X, h1s, c1s, xw1s,
                 wv + 0, wv + 128, scal[0], ytS, nullptr, xtF, actA1);
    __syncthreads();
    gemm_cell(pW1, actA1, bias1r, h1s, c1s, actA1, wave, lane);
    __syncthreads();
    // layer 2 attention (uses pre-update h2/c2) -> x_tilde2
    attention<2>(wave, lane, t, b0, X, h2s, c2s, xw2s,
                 wv + 256, wv + 384, scal[1], ytS, xtF, xt2F, actA2);
    __syncthreads();
    gemm_cell(pW2, actA2, bias2r, h2s, c2s, actA2, wave, lane);
    __syncthreads();
    // outputs: x_tilde2 -> out[0], h2 -> out[OUT1OFF]
#pragma unroll
    for (int j = 0; j < 8; ++j) {
      int idx = tid + j * 256;
      int b = idx >> 7, n = idx & 127;
      out[((size_t)(b0 + b) * TSTEPS + t) * NDIM + n] = xt2F[b * NDIM + n];
      out[OUT1OFF + ((size_t)(b0 + b) * TSTEPS + t) * HDIM + n] = h2s[b * HDIM + n];
    }
    __syncthreads();
  }
}

// ---------------------------------------------------------------------------
extern "C" void kernel_launch(void* const* d_in, const int* in_sizes, int n_in,
                              void* d_out, int out_size, void* d_ws, size_t ws_size,
                              hipStream_t stream) {
  const float* X      = (const float*)d_in[0];
  const float* y_prev = (const float*)d_in[1];
  const float* W_ih1  = (const float*)d_in[2];
  const float* W_hh1  = (const float*)d_in[3];
  const float* b_ih1  = (const float*)d_in[4];
  const float* b_hh1  = (const float*)d_in[5];
  const float* W_ih2  = (const float*)d_in[6];
  const float* W_hh2  = (const float*)d_in[7];
  const float* b_ih2  = (const float*)d_in[8];
  const float* b_hh2  = (const float*)d_in[9];
  const float* attn1_W = (const float*)d_in[10];
  const float* attn1_b = (const float*)d_in[11];
  const float* attn2_W = (const float*)d_in[12];
  const float* attn2_b = (const float*)d_in[13];

  char* ws = (char*)d_ws;
  __bf16* pW1  = (__bf16*)(ws);                    // 262144 B
  __bf16* pW2  = (__bf16*)(ws + 262144);           // 262144 B
  float*  xw1  = (float*)(ws + 524288);            // 524288 B
  float*  xw2  = (float*)(ws + 1048576);           // 524288 B
  float*  ytm  = (float*)(ws + 1572864);           // 4096 B
  float*  bs1  = (float*)(ws + 1576960);           // 2048 B
  float*  bs2  = (float*)(ws + 1579008);           // 2048 B

  prep_kernel<<<2056, 256, 0, stream>>>(
      X, y_prev, W_ih1, W_hh1, b_ih1, b_hh1, W_ih2, W_hh2, b_ih2, b_hh2,
      attn1_W, attn2_W, pW1, pW2, xw1, xw2, ytm, bs1, bs2);

  encoder_main<<<64, 256, 0, stream>>>(
      X, pW1, pW2, xw1, xw2, ytm, bs1, bs2,
      attn1_W, attn1_b, attn2_W, attn2_b, (float*)d_out);
}
